// GATLayer_44298292691034
// MI455X (gfx1250) — compile-verified, matrix operand order checked by reading, not run on hardware
//
#include <hip/hip_runtime.h>

#define ALPHA 0.2f
#define N_NODES 8192
#define D_IN 512
#define D_OUT 128
#define WORDS_PER_ROW 256  // 8192 / 32
#define M_BLK 32           // rows per attention block (2 WMMA M-tiles)

typedef __attribute__((ext_vector_type(8)))  float  v8f;
typedef __attribute__((ext_vector_type(16))) __bf16 v16bf;

union ABv { v16bf v; unsigned int u[8]; };

// Native bf16 converts: let clang emit packed v_cvt for f32->bf16 (RNE).
__device__ __forceinline__ unsigned short f2bf(float f) {
  union { __bf16 h; unsigned short u; } t;
  t.h = (__bf16)f;
  return t.u;
}
__device__ __forceinline__ unsigned int packbf2(float lo, float hi) {
  union { __bf16 h[2]; unsigned int u; } t;
  t.h[0] = (__bf16)lo;
  t.h[1] = (__bf16)hi;
  return t.u;
}
__device__ __forceinline__ float leaky(float v) { return v > 0.0f ? v : ALPHA * v; }

// ---------------------------------------------------------------------------
// Kernel 1: pack neighbor mask (int32 0/1) into bit-mask words via wave32
// ballot. 256 MB read once -> 8 MB of bits for all later passes.
// ---------------------------------------------------------------------------
__global__ void __launch_bounds__(256) pack_mask_kernel(const int* __restrict__ nb,
                                                        unsigned int* __restrict__ packed) {
  const int i    = blockIdx.x;
  const int wave = threadIdx.x >> 5;
  const int lane = threadIdx.x & 31;
  const size_t rowbase = (size_t)i * N_NODES;
  for (int t = 0; t < 32; ++t) {
    const int w = wave + t * 8;                 // 8 waves cover 256 words
    const int j = (w << 5) + lane;
    const bool pred = nb[rowbase + j] > 0;
    const unsigned int m = __builtin_amdgcn_ballot_w32(pred);
    if (lane == 0) packed[(size_t)i * WORDS_PER_ROW + w] = m;
  }
}

// ---------------------------------------------------------------------------
// Kernel 2: W (512x128 f32) -> WTb (128x512 bf16, transposed).
// ---------------------------------------------------------------------------
__global__ void __launch_bounds__(256) wt_kernel(const float* __restrict__ W,
                                                 unsigned short* __restrict__ WTb) {
  const int idx = blockIdx.x * 256 + threadIdx.x;   // 65536 total
  const int n = idx & (D_OUT - 1);
  const int k = idx >> 7;
  WTb[(size_t)n * D_IN + k] = f2bf(W[(size_t)k * D_OUT + n]);
}

// ---------------------------------------------------------------------------
// Kernel 3: h = x @ W via v_wmma_f32_16x16x32_bf16.
// ---------------------------------------------------------------------------
__global__ void __launch_bounds__(256) gemm_h_kernel(const float* __restrict__ x,
                                                     const unsigned short* __restrict__ WTb,
                                                     float* __restrict__ h,
                                                     unsigned short* __restrict__ HbT) {
  const int wave = threadIdx.x >> 5;
  const int lane = threadIdx.x & 31;
  const int lr   = lane & 15;
  const int hi   = lane >> 4;              // 0: K 0-7/16-23, 1: K 8-15/24-31
  const int m0   = blockIdx.x * 16;
  const int n0   = wave * 16;

  const float*          xrow = x   + (size_t)(m0 + lr) * D_IN;
  const unsigned short* brow = WTb + (size_t)(n0 + lr) * D_IN;

  v8f c = {0.f, 0.f, 0.f, 0.f, 0.f, 0.f, 0.f, 0.f};
  for (int kc = 0; kc < D_IN; kc += 32) {
    const int s0 = kc + (hi ? 8 : 0);
    const int s1 = s0 + 16;
    ABv a, b;
    const float4 xa0 = *(const float4*)(xrow + s0);
    const float4 xa1 = *(const float4*)(xrow + s0 + 4);
    const float4 xb0 = *(const float4*)(xrow + s1);
    const float4 xb1 = *(const float4*)(xrow + s1 + 4);
    a.u[0] = packbf2(xa0.x, xa0.y); a.u[1] = packbf2(xa0.z, xa0.w);
    a.u[2] = packbf2(xa1.x, xa1.y); a.u[3] = packbf2(xa1.z, xa1.w);
    a.u[4] = packbf2(xb0.x, xb0.y); a.u[5] = packbf2(xb0.z, xb0.w);
    a.u[6] = packbf2(xb1.x, xb1.y); a.u[7] = packbf2(xb1.z, xb1.w);
    const uint4 b0 = *(const uint4*)(brow + s0);
    const uint4 b1 = *(const uint4*)(brow + s1);
    b.u[0] = b0.x; b.u[1] = b0.y; b.u[2] = b0.z; b.u[3] = b0.w;
    b.u[4] = b1.x; b.u[5] = b1.y; b.u[6] = b1.z; b.u[7] = b1.w;
    c = __builtin_amdgcn_wmma_f32_16x16x32_bf16(false, a.v, false, b.v,
                                                (short)0, c, false, false);
  }
#pragma unroll
  for (int r = 0; r < 8; ++r) {
    const int m = m0 + r + (hi ? 8 : 0);
    const float v = c[r];
    h[(size_t)m * D_OUT + n0 + lr] = v;
    HbT[(size_t)(n0 + lr) * N_NODES + m] = f2bf(v);
  }
}

// ---------------------------------------------------------------------------
// Kernel 4: s[i] = h[i]·a_src, d[i] = h[i]·a_dst (one wave32 per row).
// ---------------------------------------------------------------------------
__global__ void __launch_bounds__(256) sd_kernel(const float* __restrict__ h,
                                                 const float* __restrict__ att,
                                                 float* __restrict__ s,
                                                 float* __restrict__ d) {
  const int wave = threadIdx.x >> 5;
  const int lane = threadIdx.x & 31;
  const int i = blockIdx.x * 8 + wave;
  const float4 hv = *(const float4*)(h + (size_t)i * D_OUT + lane * 4);
  const float4 as = *(const float4*)(att + lane * 4);
  const float4 ad = *(const float4*)(att + D_OUT + lane * 4);
  float ps = hv.x * as.x + hv.y * as.y + hv.z * as.z + hv.w * as.w;
  float pd = hv.x * ad.x + hv.y * ad.y + hv.z * ad.z + hv.w * ad.w;
#pragma unroll
  for (int off = 16; off >= 1; off >>= 1) {
    ps += __shfl_xor(ps, off, 32);
    pd += __shfl_xor(pd, off, 32);
  }
  if (lane == 0) { s[i] = ps; d[i] = pd; }
}

// ---------------------------------------------------------------------------
// Kernel 5: fused masked softmax + att @ h + ELU.
// One block = 32 output rows (2 WMMA M-tiles). The 32 KB d-vector is staged
// into LDS with CDNA5 async global->LDS copies (ASYNCcnt). Double-buffered
// P tiles: one barrier per K-chunk; each B tile feeds 2 WMMAs.
// ---------------------------------------------------------------------------
__global__ void __launch_bounds__(256) attn_kernel(const unsigned int* __restrict__ packed,
                                                   const float* __restrict__ s_g,
                                                   const float* __restrict__ d_g,
                                                   const unsigned short* __restrict__ HbT,
                                                   float* __restrict__ out) {
  __shared__ float d_lds[N_NODES];                 // 32 KB
  __shared__ float s_l[M_BLK], m_l[M_BLK], li_l[M_BLK];
  __shared__ unsigned int Pu[2][M_BLK * 16];       // double-buffered 32x32 bf16

  const int wave = threadIdx.x >> 5;
  const int lane = threadIdx.x & 31;
  const int lr   = lane & 15;
  const int hi   = lane >> 4;
  const int i0   = blockIdx.x * M_BLK;

  // ---- async global->LDS stage of d (2048 x b128 across 256 threads) ----
  for (int t = threadIdx.x; t < N_NODES / 4; t += 256) {
    const unsigned int lds_off = (unsigned int)(size_t)(&d_lds[t * 4]);
    const float* gsrc = d_g + t * 4;
    asm volatile("global_load_async_to_lds_b128 %0, %1, off"
                 :: "v"(lds_off), "v"(gsrc)
                 : "memory");
  }
  if (threadIdx.x < M_BLK) s_l[threadIdx.x] = s_g[i0 + threadIdx.x];
  asm volatile("s_wait_asynccnt 0x0" ::: "memory");
  __syncthreads();

  // ---- phase 1: per-row max then sum(exp(v - max)); 4 rows per wave ----
  for (int rr = 0; rr < 4; ++rr) {
    const int r = wave * 4 + rr;
    const float si = s_l[r];
    const unsigned int* prow = packed + (size_t)(i0 + r) * WORDS_PER_ROW;
    float mx = -3.0e38f;
    for (int w = 0; w < WORDS_PER_ROW; ++w) {
      const unsigned int word = prow[w];
      float v = leaky(si + d_lds[(w << 5) + lane]);
      v = ((word >> lane) & 1u) ? v : 0.0f;
      mx = fmaxf(mx, v);
    }
#pragma unroll
    for (int off = 16; off >= 1; off >>= 1) mx = fmaxf(mx, __shfl_xor(mx, off, 32));
    float sum = 0.0f;
    for (int w = 0; w < WORDS_PER_ROW; ++w) {
      const unsigned int word = prow[w];
      float v = leaky(si + d_lds[(w << 5) + lane]);
      v = ((word >> lane) & 1u) ? v : 0.0f;
      sum += __expf(v - mx);
    }
#pragma unroll
    for (int off = 16; off >= 1; off >>= 1) sum += __shfl_xor(sum, off, 32);
    if (lane == 0) { m_l[r] = mx; li_l[r] = 1.0f / sum; }
  }
  __syncthreads();

  // ---- phase 2: D(32x128) = P(32x8192) @ Hb(8192x128), K-chunks of 32 ----
  const int n0 = wave * 16;
  const unsigned short* brow = HbT + (size_t)(n0 + lr) * N_NODES;

  // P build: thread fills rows (tid>>4) and (tid>>4)+16, column pair (tid&15)*2
  const int pr  = threadIdx.x >> 4;
  const int pc2 = (threadIdx.x & 15) * 2;
  auto buildP = [&](int ch, int buf) {
#pragma unroll
    for (int half = 0; half < 2; ++half) {
      const int r = pr + half * 16;
      const unsigned int word = packed[(size_t)(i0 + r) * WORDS_PER_ROW + ch];
      const float m = m_l[r], li = li_l[r], si = s_l[r];
      float v0 = leaky(si + d_lds[(ch << 5) + pc2]);
      v0 = ((word >> pc2) & 1u) ? v0 : 0.0f;
      float v1 = leaky(si + d_lds[(ch << 5) + pc2 + 1]);
      v1 = ((word >> (pc2 + 1)) & 1u) ? v1 : 0.0f;
      Pu[buf][r * 16 + (pc2 >> 1)] = packbf2(__expf(v0 - m) * li, __expf(v1 - m) * li);
    }
  };

  v8f c0 = {0.f, 0.f, 0.f, 0.f, 0.f, 0.f, 0.f, 0.f};
  v8f c1 = {0.f, 0.f, 0.f, 0.f, 0.f, 0.f, 0.f, 0.f};
  const int asel = hi ? 4 : 0;

  buildP(0, 0);
  for (int ch = 0; ch < WORDS_PER_ROW; ++ch) {
    __syncthreads();                          // P[buf] ready for all waves
    const int buf = ch & 1;

    const int s0 = (ch << 5) + (hi ? 8 : 0);
    const uint4 b0 = *(const uint4*)(brow + s0);
    const uint4 b1 = *(const uint4*)(brow + s0 + 16);
    __builtin_prefetch((const void*)(brow + s0 + 512), 0, 1);  // ~16 chunks ahead

    ABv a0, a1, b;
    b.u[0] = b0.x; b.u[1] = b0.y; b.u[2] = b0.z; b.u[3] = b0.w;
    b.u[4] = b1.x; b.u[5] = b1.y; b.u[6] = b1.z; b.u[7] = b1.w;
#pragma unroll
    for (int t = 0; t < 4; ++t) {
      a0.u[t]     = Pu[buf][lr * 16 + asel + t];
      a0.u[4 + t] = Pu[buf][lr * 16 + asel + 8 + t];
      a1.u[t]     = Pu[buf][(lr + 16) * 16 + asel + t];
      a1.u[4 + t] = Pu[buf][(lr + 16) * 16 + asel + 8 + t];
    }
    c0 = __builtin_amdgcn_wmma_f32_16x16x32_bf16(false, a0.v, false, b.v,
                                                 (short)0, c0, false, false);
    c1 = __builtin_amdgcn_wmma_f32_16x16x32_bf16(false, a1.v, false, b.v,
                                                 (short)0, c1, false, false);
    if (ch + 1 < WORDS_PER_ROW) buildP(ch + 1, buf ^ 1);  // overlap with WMMA reads
  }

  // ---- epilogue: ELU + store (rows i0.. for c0, i0+16.. for c1) ----
#pragma unroll
  for (int r = 0; r < 8; ++r) {
    const int m = i0 + r + (hi ? 8 : 0);
    float v = c0[r];
    v = v > 0.0f ? v : (__expf(v) - 1.0f);
    out[(size_t)m * D_OUT + n0 + lr] = v;

    const int m2 = m + 16;
    float w = c1[r];
    w = w > 0.0f ? w : (__expf(w) - 1.0f);
    out[(size_t)m2 * D_OUT + n0 + lr] = w;
  }
}

// ---------------------------------------------------------------------------
extern "C" void kernel_launch(void* const* d_in, const int* in_sizes, int n_in,
                              void* d_out, int out_size, void* d_ws, size_t ws_size,
                              hipStream_t stream) {
  const float* x   = (const float*)d_in[0];
  const int*   nb  = (const int*)d_in[1];
  const float* w   = (const float*)d_in[2];
  const float* att = (const float*)d_in[3];
  float* out = (float*)d_out;

  // workspace layout (~14.3 MB)
  char* ws = (char*)d_ws;
  unsigned int*   packed = (unsigned int*)ws;                         // 8 MB
  float*          h      = (float*)(ws + (8u << 20));                 // 4 MB
  unsigned short* HbT    = (unsigned short*)(ws + (12u << 20));       // 2 MB
  unsigned short* WTb    = (unsigned short*)(ws + (14u << 20));       // 128 KB
  float*          s      = (float*)(ws + (14u << 20) + (128u << 10)); // 32 KB
  float*          d      = s + N_NODES;                               // 32 KB

  pack_mask_kernel<<<N_NODES, 256, 0, stream>>>(nb, packed);
  wt_kernel<<<(D_IN * D_OUT) / 256, 256, 0, stream>>>(w, WTb);
  gemm_h_kernel<<<N_NODES / 16, 256, 0, stream>>>(x, WTb, h, HbT);
  sd_kernel<<<N_NODES / 8, 256, 0, stream>>>(h, att, s, d);
  attn_kernel<<<N_NODES / M_BLK, 256, 0, stream>>>(packed, s, d, HbT, out);
}
